// _DepthwiseMultiScaleConv_31877247271456
// MI455X (gfx1250) — matrix operand order repaired and checked by reading, not compiled, not run on hardware
//
#include <hip/hip_runtime.h>
#include <hip/hip_bf16.h>
#include <stdint.h>

// Problem constants (from reference): B=2, L=4096, H=16, D=128
#define B_N     2
#define L_LEN   4096
#define H_N     16
#define D_N     128
#define C_N     2048        // H*D depthwise channels
#define M_N     640         // 5*D mixed features per head
#define KTILES  20          // 640 / 32 (bf16 WMMA K)
#define NTILES  8           // 128 / 16
#define MT_N    8           // M-tiles (of 16 rows) per workgroup -> 128 l-rows
#define YSTRIDE 648         // 640 + 8 pad (ushort) -> row stride 1296B, conflict-free A reads
#define WSTRIDE 33          // tap LDS stride (f32) to spread banks

#define WS_FRAGS (NTILES * KTILES * 32)        // 5120 fragment-lanes, 32B each
#define B_BYTES  (WS_FRAGS * 32)               // 163840 B of bf16 B-fragments
#define Y_BYTES  (16 * YSTRIDE * 2)            // 20736 B
#define W_BYTES  (M_N * WSTRIDE * 4)           // 84480 B of f32 taps
#define SMEM_BYTES (B_BYTES + Y_BYTES + W_BYTES)   // 269056 B < 320KB WGP LDS

typedef __attribute__((ext_vector_type(16))) __bf16   v16bf;
typedef __attribute__((ext_vector_type(8)))  float    v8f;
typedef __attribute__((ext_vector_type(4)))  uint32_t v4u;
typedef __attribute__((ext_vector_type(4)))  uint32_t u32x4;
typedef __attribute__((ext_vector_type(8)))  int32_t  i32x8;
typedef __attribute__((ext_vector_type(4)))  int32_t  i32x4;

union FragBF { v16bf bf; v4u u2[2]; };

static __device__ __forceinline__ uint16_t f32_to_bf16(float f) {
  union { float f; uint32_t u; } v; v.f = f;
  uint32_t r = v.u + 0x7FFFu + ((v.u >> 16) & 1u);   // round-to-nearest-even
  return (uint16_t)(r >> 16);
}

// ---------------------------------------------------------------------------
// Prep: swizzle w_mix [128 x 640] f32 into fragment-ready bf16 B-tiles.
// B (32x16, KxN) in contiguous k order, same as A here: lanes 0-15 -> N=lane, elem e -> K=e;
// lanes 16-31 -> N=lane-16, elem e -> K=16+e. Stored [t][s][lane][16] in d_ws.
// ---------------------------------------------------------------------------
__global__ void wmix_prep_kernel(const float* __restrict__ wmix,
                                 uint16_t* __restrict__ wsB) {
  int idx = blockIdx.x * 256 + threadIdx.x;
  if (idx >= WS_FRAGS) return;
  int lane = idx & 31;
  int s    = (idx >> 5) % KTILES;
  int t    = idx / (32 * KTILES);
  int n    = t * 16 + (lane & 15);
  int hlf  = lane >> 4;
  uint16_t* dst = wsB + (size_t)idx * 16;
  const float* src = wmix + (size_t)n * M_N + s * 32 + hlf * 16;
#pragma unroll
  for (int e = 0; e < 16; ++e) dst[e] = f32_to_bf16(src[e]);
}

// ---------------------------------------------------------------------------
// Depthwise causal conv for one channel column. K compile-time: taps read from
// LDS, x sliding window lives in VGPRs. y[l] = sum_j w[j]*x[l + j - (K-1)].
// ---------------------------------------------------------------------------
template <int K>
static __device__ __forceinline__ void conv_col(const float* __restrict__ xp,  // &x[b][0][c]
                                                const float* wt_lds,           // &lds_w[m*WSTRIDE]
                                                int l0,
                                                uint16_t* ydst) {              // &lds_y[m]
  float wt[K];
#pragma unroll
  for (int j = 0; j < K; ++j) wt[j] = wt_lds[j];
  float xv[15 + K];
#pragma unroll
  for (int r = 0; r < 15 + K; ++r) {
    int l = l0 - (K - 1) + r;
    xv[r] = (l >= 0) ? xp[(size_t)l * C_N] : 0.0f;
  }
#pragma unroll
  for (int li = 0; li < 16; ++li) {
    float s = 0.0f;
#pragma unroll
    for (int j = 0; j < K; ++j) s = fmaf(wt[j], xv[li + j], s);
    ydst[li * YSTRIDE] = f32_to_bf16(s);
  }
}

// ---------------------------------------------------------------------------
// Fused kernel v2: one workgroup = (b, head, 128-row l-group = 8 M-tiles).
//  * B fragments (160KB) DMA'd global->LDS ONCE per WG by the Tensor Data
//    Mover (wave 0 issues tensor_load_to_lds, overlapped with first conv,
//    completion via s_wait_tensorcnt).  Cuts B L2 traffic 8x.
//  * Conv taps staged to LDS once per WG.
//  * Per M-tile: 256 threads conv 16x640 Y tile (bf16, LDS), then 8 waves x
//    20x v_wmma_f32_16x16x32_bf16 (one 16-wide N-tile each).
// ---------------------------------------------------------------------------
__global__ void __launch_bounds__(256, 1)
conv_mix_kernel(const float* __restrict__ x,
                const float* __restrict__ w1, const float* __restrict__ w2,
                const float* __restrict__ w3, const float* __restrict__ w4,
                const float* __restrict__ w5,
                const uint16_t* __restrict__ wsB,
                float* __restrict__ out) {
  extern __shared__ __align__(16) char smem[];
  uint16_t* lds_B = (uint16_t*)smem;                       // offset 0 (matches D# lds_addr)
  uint16_t* lds_y = (uint16_t*)(smem + B_BYTES);
  float*    lds_w = (float*)(smem + B_BYTES + Y_BYTES);

  const int tid  = threadIdx.x;
  const int wv   = tid >> 5;
  const int lane = tid & 31;

  const int tile = blockIdx.x;                 // 1024 blocks
  const int h    = tile & (H_N - 1);
  const int lg   = (tile >> 4) & 31;           // L/128 = 32 groups
  const int b    = tile >> 9;
  const int lbase = lg * (MT_N * 16);

  const float* xb = x + (size_t)b * L_LEN * C_N;

  // ---- Wave 0: kick off TDM DMA of the 160KB B-fragment block into LDS ----
  if (wv == 0) {
    const uint64_t ga = (uint64_t)(uintptr_t)wsB;
    // D# group0: count=1, lds_addr=0, global_addr, type=2
    u32x4 g0 = { 1u, 0u, (uint32_t)ga, (uint32_t)(ga >> 32) | 0x80000000u };
    // D# group1: data_size=3 (8B), tensor_dim0 = tile_dim0 = 20480 (8B units),
    //            tensor_dim1 = tile_dim1 = 1, stride0 = 20480
    i32x8 g1 = { 0x30000, 0x50000000, 0x00010000, 0x50000000,
                 1, 20480, 0, 0 };
    i32x4 gz  = { 0, 0, 0, 0 };
    i32x8 gz8 = { 0, 0, 0, 0, 0, 0, 0, 0 };
    __builtin_amdgcn_tensor_load_to_lds(g0, g1, gz, gz, gz8, 0);
  }

  // ---- Stage conv taps into LDS (once per WG) ----
  for (int m = tid; m < M_N; m += 256) {
    const int f  = h * M_N + m;
    const int ki = f >> 11;
    const int c  = f & (C_N - 1);
    const float* src; int k;
    switch (ki) {
      case 0:  src = w1 + (size_t)c * 1;  k = 1;  break;
      case 1:  src = w2 + (size_t)c * 3;  k = 3;  break;
      case 2:  src = w3 + (size_t)c * 7;  k = 7;  break;
      case 3:  src = w4 + (size_t)c * 15; k = 15; break;
      default: src = w5 + (size_t)c * 31; k = 31; break;
    }
    float* dst = lds_w + m * WSTRIDE;
    for (int j = 0; j < k; ++j) dst[j] = src[j];
  }
  __syncthreads();    // taps ready

  // ---- First conv tile (overlaps with the TDM transfer) ----
  {
    const int l0 = lbase;
    for (int m = tid; m < M_N; m += 256) {
      const int f  = h * M_N + m;
      const int ki = f >> 11;
      const int c  = f & (C_N - 1);
      const float* xp = xb + c;
      const float* wt = lds_w + m * WSTRIDE;
      uint16_t* yd = lds_y + m;
      switch (ki) {
        case 0:  conv_col<1 >(xp, wt, l0, yd); break;
        case 1:  conv_col<3 >(xp, wt, l0, yd); break;
        case 2:  conv_col<7 >(xp, wt, l0, yd); break;
        case 3:  conv_col<15>(xp, wt, l0, yd); break;
        default: conv_col<31>(xp, wt, l0, yd); break;
      }
    }
  }
  if (wv == 0) __builtin_amdgcn_s_wait_tensorcnt(0);   // B fragments landed
  __syncthreads();    // Y tile 0 + B ready

  const int hlf = lane >> 4;
  const int row = lane & 15;
  const uint16_t* bfrag_base = lds_B + ((size_t)wv * KTILES * 32 + lane) * 16;

  for (int mt = 0; mt < MT_N; ++mt) {
    const int l0 = lbase + mt * 16;

    // ---- WMMA: A from lds_y, B from lds_B ----
    v8f acc = {};
    for (int s = 0; s < KTILES; ++s) {
      FragBF a;
      const v4u* ap = (const v4u*)&lds_y[row * YSTRIDE + s * 32 + hlf * 16];
      a.u2[0] = ap[0];
      a.u2[1] = ap[1];                       // +8 ushorts
      FragBF bm;
      const v4u* bp = (const v4u*)(bfrag_base + (size_t)s * 32 * 16);
      bm.u2[0] = bp[0];
      bm.u2[1] = bp[1];
      acc = __builtin_amdgcn_wmma_f32_16x16x32_bf16(
          false, a.bf, false, bm.bf, (short)0, acc, false, false);
    }

    // ---- Store: C/D layout VGPR r -> M = r + 8*hlf, N = lane&15 ----
    const int n = wv * 16 + row;
    const size_t obase = (((size_t)b * L_LEN + l0) * H_N + h) * D_N + n;
#pragma unroll
    for (int r = 0; r < 8; ++r) {
      const int M = hlf * 8 + r;
      out[obase + (size_t)M * (H_N * D_N)] = acc[r];
    }

    // ---- Next conv tile into lds_y ----
    if (mt + 1 < MT_N) {
      __syncthreads();    // everyone done reading lds_y
      const int l1 = lbase + (mt + 1) * 16;
      for (int m = tid; m < M_N; m += 256) {
        const int f  = h * M_N + m;
        const int ki = f >> 11;
        const int c  = f & (C_N - 1);
        const float* xp = xb + c;
        const float* wt = lds_w + m * WSTRIDE;
        uint16_t* yd = lds_y + m;
        switch (ki) {
          case 0:  conv_col<1 >(xp, wt, l1, yd); break;
          case 1:  conv_col<3 >(xp, wt, l1, yd); break;
          case 2:  conv_col<7 >(xp, wt, l1, yd); break;
          case 3:  conv_col<15>(xp, wt, l1, yd); break;
          default: conv_col<31>(xp, wt, l1, yd); break;
        }
      }
      __syncthreads();    // Y tile mt+1 ready
    }
  }
}

extern "C" void kernel_launch(void* const* d_in, const int* in_sizes, int n_in,
                              void* d_out, int out_size, void* d_ws, size_t ws_size,
                              hipStream_t stream) {
  const float* x    = (const float*)d_in[0];
  const float* w1   = (const float*)d_in[1];
  const float* w2   = (const float*)d_in[2];
  const float* w3   = (const float*)d_in[3];
  const float* w4   = (const float*)d_in[4];
  const float* w5   = (const float*)d_in[5];
  const float* wmix = (const float*)d_in[6];
  float*    out = (float*)d_out;
  uint16_t* wsB = (uint16_t*)d_ws;   // needs 160KB scratch

  // Pre-swizzle w_mix into bf16 WMMA-B fragments (rewritten every call).
  wmix_prep_kernel<<<(WS_FRAGS + 255) / 256, 256, 0, stream>>>(wmix, wsB);

  // One workgroup per (b, 128-row l-group, head): 2 * 32 * 16 = 1024 blocks.
  const int nblocks = B_N * (L_LEN / (MT_N * 16)) * H_N;
  conv_mix_kernel<<<nblocks, 256, SMEM_BYTES, stream>>>(x, w1, w2, w3, w4, w5,
                                                        wsB, out);
}
